// UAVAuctionModel_16063177687588
// MI455X (gfx1250) — matrix operand order repaired and checked by reading, not compiled, not run on hardware
//
#include <hip/hip_runtime.h>

typedef __attribute__((ext_vector_type(16))) _Float16 v16h;
typedef __attribute__((ext_vector_type(8)))  float    v8f;

#define BATCH   32768
#define NNODES  64
#define TOT     (BATCH * NNODES)
#define BPB     8                       // batches per block
#define NBLK    (BATCH / BPB)           // 4096 blocks

// D^ETA * (1+THETA) = 5^0.1 * 1.1
#define C_REWARD 1.29208083748817f

__global__ __launch_bounds__(128) void uav_auction_kernel(
    const float* __restrict__ sr_g, const float* __restrict__ te_g,
    const float* __restrict__ re_g, const float* __restrict__ w1_g,
    const float* __restrict__ b1_g, const float* __restrict__ w2_g,
    const float* __restrict__ b2_g, const float* __restrict__ w3_g,
    const float* __restrict__ b3_g, float* __restrict__ out)
{
    __shared__ float s_sr[BPB * 64];
    __shared__ float s_val[BPB * 64];
    __shared__ float s_vv[64];
    __shared__ float s_tot[BPB];
    __shared__ float s_w1[128];
    __shared__ float s_b1[64];
    __shared__ float s_b2[64];
    __shared__ float s_w3[64];
    __shared__ __align__(32) _Float16 s_w2t[64 * 64];   // transposed [n][k], f16

    const int tid  = threadIdx.x;
    const int lane = tid & 31;
    const int wv   = tid >> 5;
    const int base = blockIdx.x * (BPB * 64);           // flat element base

    // ---- stage weights to LDS (amortized over BPB batches) ----
    s_w1[tid] = w1_g[tid];                               // W1 is 2x64 = 128
    if (tid < 64) {
        s_b1[tid] = b1_g[tid];
        s_b2[tid] = b2_g[tid];
        s_w3[tid] = w3_g[tid];
    }
    // W2 (64x64 f32 row-major [k][n]) -> LDS f16 transposed [n][k]
    #pragma unroll
    for (int i = 0; i < 32; ++i) {
        int e = i * 128 + tid;                           // coalesced read
        s_w2t[(e & 63) * 64 + (e >> 6)] = (_Float16)w2_g[e];
    }

    // ---- stage inputs for all BPB batches (fully coalesced, 4/thread) ----
    float te[4], re[4];
    #pragma unroll
    for (int i = 0; i < 4; ++i) {
        int e = i * 128 + tid;
        s_sr[e] = sr_g[base + e];
        te[i]   = te_g[base + e];
        re[i]   = re_g[base + e];
    }
    __syncthreads();

    // ---- per-batch sensing totals ----
    if (tid < BPB) {
        float tot = 0.0f;
        #pragma unroll
        for (int i = 0; i < 64; ++i) tot += s_sr[tid * 64 + i];
        s_tot[tid] = tot;
    }
    __syncthreads();

    // ---- valuations (alpha=0.5 -> 2*sqrt(1+eff)) for all BPB batches ----
    #pragma unroll
    for (int i = 0; i < 4; ++i) {
        int e = i * 128 + tid;
        float sr  = s_sr[e];
        float eff = C_REWARD * sr / s_tot[e >> 6] * (te[i] / re[i]);
        float val = 2.0f * sqrtf(1.0f + eff);
        s_val[e] = val;
        out[2 * TOT + base + e] = val;                   // valuations
    }
    __syncthreads();

    const float b3v  = b3_g[0];
    const int   hi   = lane >> 4;                        // K-half select
    const int   col0 = lane & 15;
    const int   mr   = wv * 16 + col0;                   // row within batch

    #pragma unroll 1
    for (int bi = 0; bi < BPB; ++bi) {
        const int bb = base + bi * 64;                   // this batch's flat base
        const float vm = s_val[bi * 64 + mr];
        const float sm = s_sr[bi * 64 + mr];

        // ---- layer 1 fused into A fragments, layer 2 via WMMA ----
        v8f c[4] = {};                                   // 4 N-tiles of 16x16 f32
        #pragma unroll
        for (int q = 0; q < 2; ++q) {                    // two K=32 chunks
            // A layout: low lanes K = jb..jb+15 ; high lanes +16
            v16h a;
            const int jb = q * 32 + hi * 16;
            #pragma unroll
            for (int r = 0; r < 8; ++r) {
                int j = jb + 2 * r;
                float h0 = fmaxf(vm * s_w1[j]     + sm * s_w1[64 + j]     + s_b1[j],     0.0f);
                float h1 = fmaxf(vm * s_w1[j + 1] + sm * s_w1[64 + j + 1] + s_b1[j + 1], 0.0f);
                a[2 * r]     = (_Float16)h0;
                a[2 * r + 1] = (_Float16)h1;
            }
            #pragma unroll
            for (int t2 = 0; t2 < 4; ++t2) {
                // B fragment: column n2, contiguous K run (low lanes K=q*32..+15, high +16)
                const int n2 = t2 * 16 + col0;
                const v16h bf = *(const v16h*)&s_w2t[n2 * 64 + q * 32 + hi * 16];
                c[t2] = __builtin_amdgcn_wmma_f32_16x16x32_f16(
                            false, a, false, bf, (short)0, c[t2], false, false);
            }
        }

        // ---- bias + relu + layer 3 dot, fused on C fragments ----
        #pragma unroll
        for (int vr = 0; vr < 8; ++vr) {                 // C VGPR -> row vr + 8*hi
            float acc = 0.0f;
            #pragma unroll
            for (int t2 = 0; t2 < 4; ++t2) {
                int cidx = t2 * 16 + col0;               // hidden unit index
                acc += fmaxf(c[t2][vr] + s_b2[cidx], 0.0f) * s_w3[cidx];
            }
            // reduce over 16 lanes; xor 1/2/4/8 stays inside each half-wave
            acc += __shfl_xor(acc, 1, 32);
            acc += __shfl_xor(acc, 2, 32);
            acc += __shfl_xor(acc, 4, 32);
            acc += __shfl_xor(acc, 8, 32);
            if (col0 == 0) {
                int row  = wv * 16 + hi * 8 + vr;
                float vv = acc + b3v;
                s_vv[row] = vv;
                out[3 * TOT + bb + row] = vv;            // virtual_values
            }
        }
        __syncthreads();

        // ---- argmax / second-max -> allocations & payments ----
        if (tid < 64) {
            float m1 = -3.402823466e38f, m2 = -3.402823466e38f;
            int   i1 = 0;
            #pragma unroll 4
            for (int n = 0; n < 64; ++n) {
                float x = s_vv[n];
                if (x > m1)      { m2 = m1; m1 = x; i1 = n; }  // first max wins
                else if (x > m2) { m2 = x; }
            }
            out[bb + tid]       = (tid == i1) ? 1.0f : 0.0f;          // allocations
            out[TOT + bb + tid] = (tid == i1) ? fmaxf(m2, 0.0f) : 0.0f; // payments
        }
        __syncthreads();                                 // s_vv reused next batch
    }
}

extern "C" void kernel_launch(void* const* d_in, const int* in_sizes, int n_in,
                              void* d_out, int out_size, void* d_ws, size_t ws_size,
                              hipStream_t stream) {
    const float* sr = (const float*)d_in[0];
    const float* te = (const float*)d_in[1];
    const float* re = (const float*)d_in[2];
    const float* w1 = (const float*)d_in[3];
    const float* b1 = (const float*)d_in[4];
    const float* w2 = (const float*)d_in[5];
    const float* b2 = (const float*)d_in[6];
    const float* w3 = (const float*)d_in[7];
    const float* b3 = (const float*)d_in[8];
    float* out = (float*)d_out;

    uav_auction_kernel<<<dim3(NBLK), dim3(128), 0, stream>>>(
        sr, te, re, w1, b1, w2, b2, w3, b3, out);
}
